// relative_position_aware_multi_head_attention_69518340653262
// MI455X (gfx1250) — matrix operand order repaired and checked by reading, not compiled, not run on hardware
//
#include <hip/hip_runtime.h>
#include <hip/hip_bf16.h>
#include <stdint.h>

// ---------------------------------------------------------------------------
// Relative-position multi-head attention for MI455X (gfx1250), bf16 WMMA path
// B=8, S=1024, D=1024, H=16, dk=64
// ---------------------------------------------------------------------------

typedef __attribute__((ext_vector_type(16))) __bf16 v16bf;
typedef __attribute__((ext_vector_type(8)))  float  v8f;
typedef __attribute__((ext_vector_type(4)))  int    v4i;
typedef __attribute__((ext_vector_type(8)))  int    v8i;

#define SEQ    1024
#define DMODEL 1024
#define NHEAD  16
#define DK     64
#define BATCH  8
#define BH     (BATCH * NHEAD)

union BF16x16 {
    v16bf v;
    uint4 q[2];
    unsigned short u[16];
};

__device__ __forceinline__ unsigned short f2bf(float f) {
    // native convert -> hardware bf16 cvt op
    union { __bf16 h; unsigned short u; } x;
    x.h = (__bf16)f;
    return x.u;
}
__device__ __forceinline__ float bf2f(unsigned short h) {
    union { unsigned u; float f; } x; x.u = ((unsigned)h) << 16; return x.f;
}
__device__ __forceinline__ v8f wmma_bf16(v16bf a, v16bf b, v8f c) {
    // D = A(16x32 bf16) * B(32x16 bf16) + C(16x16 f32)
    return __builtin_amdgcn_wmma_f32_16x16x32_bf16(false, a, false, b,
                                                   (short)0, c, false, false);
}

// ---------------------------------------------------------------------------
// Kernel 1: projection GEMM  Y = X @ W + b  (fp32 in, bf16 out)
//   grid (64 m-blocks, 16 n-blocks), 256 threads (8 waves)
//   block tile: M=128, N=64, K-step=32
//   vlayout==0 : out[b,h,s,d]  (Q / K)       vlayout==1 : out[b,h,d,s]  (V^T)
//   oscale     : 0.125 for K (folds 1/sqrt(dk) into K), else 1.0
// ---------------------------------------------------------------------------
__global__ __launch_bounds__(256) void proj_gemm_kernel(
    const float* __restrict__ X, const float* __restrict__ W,
    const float* __restrict__ bias, unsigned short* __restrict__ out,
    int vlayout, float oscale)
{
    __shared__ __align__(16) unsigned short As[128][32];  // A-matrix tile (bf16)
    __shared__ __align__(16) unsigned short Bt[64][32];   // B tile, transposed

    const int tid = threadIdx.x;
    const int w = tid >> 5, L = tid & 31, lo = L & 15, hi = L >> 4;
    const int mb = blockIdx.x, nb = blockIdx.y;
    const int mt0 = 2 * (w & 3), nt0 = 2 * (w >> 2);   // 2x2 wmma tiles per wave

    v8f acc[2][2];
    for (int i = 0; i < 2; ++i)
        for (int j = 0; j < 2; ++j)
            for (int r = 0; r < 8; ++r) acc[i][j][r] = 0.f;

    for (int ks = 0; ks < 32; ++ks) {
        const int k0 = ks * 32;
        // stage X[128 x 32] -> As (coalesced along k, convert fp32->bf16)
        #pragma unroll
        for (int i = 0; i < 16; ++i) {
            int e = tid + i * 256;
            int row = e >> 5, col = e & 31;
            As[row][col] = f2bf(X[(size_t)(mb * 128 + row) * DMODEL + k0 + col]);
        }
        // stage W[32 x 64] -> Bt[64][32] transposed (coalesced along n)
        #pragma unroll
        for (int i = 0; i < 8; ++i) {
            int e = tid + i * 256;
            int kk = e >> 6, c = e & 63;
            Bt[c][kk] = f2bf(W[(size_t)(k0 + kk) * DMODEL + nb * 64 + c]);
        }
        __syncthreads();

        BF16x16 a[2], b[2];
        #pragma unroll
        for (int im = 0; im < 2; ++im) {
            int m = (mt0 + im) * 16 + lo;           // A: lane holds row m
            a[im].q[0] = *(const uint4*)&As[m][hi * 16];       // K = 16*hi..+7
            a[im].q[1] = *(const uint4*)&As[m][hi * 16 + 8];   // K = 16*hi+8..+15
        }
        #pragma unroll
        for (int in = 0; in < 2; ++in) {
            int n = (nt0 + in) * 16 + lo;           // B: lane holds column n
            b[in].q[0] = *(const uint4*)&Bt[n][hi * 16];       // K = 16*hi..+15
            b[in].q[1] = *(const uint4*)&Bt[n][hi * 16 + 8];
        }
        #pragma unroll
        for (int im = 0; im < 2; ++im)
            #pragma unroll
            for (int in = 0; in < 2; ++in)
                acc[im][in] = wmma_bf16(a[im].v, b[in].v, acc[im][in]);
        __syncthreads();
    }

    // epilogue: bias, scale, layout transform, bf16 store
    for (int im = 0; im < 2; ++im)
        for (int in = 0; in < 2; ++in) {
            int C = nb * 64 + (nt0 + in) * 16 + lo;
            float bv = bias[C];
            int hcol = C >> 6, d = C & 63;
            #pragma unroll
            for (int r = 0; r < 8; ++r) {
                int R = mb * 128 + (mt0 + im) * 16 + r + 8 * hi;
                int bb = R >> 10, s = R & 1023;
                float val = (acc[im][in][r] + bv) * oscale;
                size_t addr = vlayout
                    ? ((size_t)(bb * NHEAD + hcol) * DK + d) * SEQ + s
                    : ((size_t)(bb * NHEAD + hcol) * SEQ + s) * DK + d;
                out[addr] = f2bf(val);
            }
        }
}

// ---------------------------------------------------------------------------
// Kernel 2: E2[b,h,q,k] = sum_d Q[b,h,q,d] * rel[q,k,d]
//   restructured per-q GEMM: [128(bh) x 64] @ [64 x 1024], bf16 out
//   grid (16 k-chunks, 1024 q), 256 threads; reads rel exactly once.
//   Q tile (128 rows x 64 bf16, row stride SEQ*DK) staged to LDS by the
//   Tensor Data Mover (tensor_load_to_lds, D# per ISA 08_async_tensor §8).
// ---------------------------------------------------------------------------
__global__ __launch_bounds__(256) void e2_kernel(
    const unsigned short* __restrict__ Qbf, const float* __restrict__ rel,
    unsigned short* __restrict__ E2)
{
    __shared__ __align__(16) unsigned short As[128 * 64];  // Q[:,q,:] 16 KB
    const int tid = threadIdx.x;
    const int nchunk = blockIdx.x, q = blockIdx.y;
    const int w = tid >> 5, L = tid & 31, lo = L & 15, hi = L >> 4;

    // --- TDM: DMA the 2D tile Qbf[row, q*64 .. q*64+63], row = 0..127,
    //     row stride SEQ*DK elements, into LDS. One wave issues.
    if (w == 0) {
        unsigned lds_off = (unsigned)(uintptr_t)(void*)&As[0];
        unsigned long long gaddr =
            (unsigned long long)(uintptr_t)(Qbf + (size_t)q * DK);
        v4i g0;
        g0[0] = 0x1;                                   // count=1, user mode
        g0[1] = (int)lds_off;                          // lds_addr (bytes)
        g0[2] = (int)(unsigned)(gaddr & 0xFFFFFFFFull);          // global_addr lo
        g0[3] = (int)((unsigned)((gaddr >> 32) & 0x1FFFFFFull)   // global_addr hi
                      | (2u << 30));                   // type=2 ("image")
        v8i g1;
        g1[0] = 0x00010000;          // workgroup_mask=0, data_size=1 (2 bytes)
        g1[1] = (int)(64u << 16);    // tensor_dim0 = 64  (bits 63:48)
        g1[2] = (int)(128u << 16);   // tensor_dim1 = 128 (bits 95:80)
        g1[3] = (int)(64u << 16);    // tile_dim0  = 64   (bits 127:112)
        g1[4] = 128;                 // tile_dim1  = 128  (bits 143:128)
        g1[5] = (int)(SEQ * DK);     // tensor_dim0_stride = 65536 (bits 191:160)
        g1[6] = 0;                   // stride hi / dim1_stride lo
        g1[7] = 0;                   // dim1_stride hi
        asm volatile("tensor_load_to_lds %0, %1" :: "s"(g0), "s"(g1) : "memory");
        __builtin_amdgcn_s_wait_tensorcnt(0);
    }
    __syncthreads();

    BF16x16 a0, a1;                             // A rows = bh (wave's m-tile)
    {
        const unsigned short* Ar = As + (w * 16 + lo) * 64;
        a0.q[0] = *(const uint4*)(Ar + hi * 16);
        a0.q[1] = *(const uint4*)(Ar + hi * 16 + 8);
        a1.q[0] = *(const uint4*)(Ar + 32 + hi * 16);
        a1.q[1] = *(const uint4*)(Ar + 32 + hi * 16 + 8);
    }

    for (int j = 0; j < 4; ++j) {
        int k0 = nchunk * 64 + j * 16;
        const float* rp = rel + ((size_t)q * SEQ + (k0 + lo)) * DK; // d-contig
        BF16x16 b0, b1;
        #pragma unroll
        for (int t = 0; t < 4; ++t) {           // d = 16*hi .. +15
            float4 f = ((const float4*)(rp + hi * 16))[t];
            b0.u[4 * t + 0] = f2bf(f.x); b0.u[4 * t + 1] = f2bf(f.y);
            b0.u[4 * t + 2] = f2bf(f.z); b0.u[4 * t + 3] = f2bf(f.w);
        }
        #pragma unroll
        for (int t = 0; t < 4; ++t) {           // d = 32 + 16*hi .. +15
            float4 f = ((const float4*)(rp + 32 + hi * 16))[t];
            b1.u[4 * t + 0] = f2bf(f.x); b1.u[4 * t + 1] = f2bf(f.y);
            b1.u[4 * t + 2] = f2bf(f.z); b1.u[4 * t + 3] = f2bf(f.w);
        }
        v8f c; for (int r = 0; r < 8; ++r) c[r] = 0.f;
        c = wmma_bf16(a0.v, b0.v, c);
        c = wmma_bf16(a1.v, b1.v, c);
        #pragma unroll
        for (int r = 0; r < 8; ++r) {
            int bh = w * 16 + r + 8 * hi;
            E2[(size_t)bh * SEQ * SEQ + (size_t)q * SEQ + k0 + lo] = f2bf(c[r]);
        }
    }
}

// ---------------------------------------------------------------------------
// Kernel 3: flash attention. One wave per (b,h,16-row q tile); online softmax
//   over k in steps of 32. Scores C-operand is initialized with the E2 tile.
// ---------------------------------------------------------------------------
__global__ __launch_bounds__(256) void attn_kernel(
    const unsigned short* __restrict__ Qbf, const unsigned short* __restrict__ Kbf,
    const unsigned short* __restrict__ Vt,  const unsigned short* __restrict__ E2,
    const unsigned char* __restrict__ mask, float* __restrict__ out)
{
    __shared__ __align__(16) unsigned short Plds[8][16][32];  // per-wave P tile

    const int tid = threadIdx.x;
    const int w = tid >> 5, L = tid & 31, lo = L & 15, hi = L >> 4;
    const int g = blockIdx.x * 8 + w;
    const int qt = g & 63, bh = g >> 6;
    const int b = bh >> 4, h = bh & 15;
    const int q0 = qt * 16;

    BF16x16 a0, a1;                         // Q tile in A-matrix layout
    {
        const unsigned short* Qr = Qbf + ((size_t)bh * SEQ + q0 + lo) * DK;
        a0.q[0] = *(const uint4*)(Qr + hi * 16);
        a0.q[1] = *(const uint4*)(Qr + hi * 16 + 8);
        a1.q[0] = *(const uint4*)(Qr + 32 + hi * 16);
        a1.q[1] = *(const uint4*)(Qr + 32 + hi * 16 + 8);
    }

    v8f o[4];
    for (int t = 0; t < 4; ++t) for (int r = 0; r < 8; ++r) o[t][r] = 0.f;
    float mrow[8], lrow[8];
    for (int r = 0; r < 8; ++r) { mrow[r] = -3.0e38f; lrow[r] = 0.f; }

    const size_t e2base = (size_t)bh * SEQ * SEQ;
    const size_t mbase  = (size_t)b  * SEQ * SEQ;

    for (int kt = 0; kt < 32; ++kt) {
        const int kbase = kt * 32;
        // prefetch next K/V tiles (global_prefetch_b8)
        if (kt + 1 < 32) {
            __builtin_prefetch(Kbf + ((size_t)bh * SEQ + kbase + 32 + lo) * DK, 0, 0);
            __builtin_prefetch(Vt + ((size_t)bh * DK + lo) * SEQ + kbase + 32, 0, 0);
        }
        float s[2][8];
        #pragma unroll
        for (int sub = 0; sub < 2; ++sub) {
            const int key = kbase + sub * 16 + lo;
            v8f c;                               // C init = E2 tile (free add)
            #pragma unroll
            for (int r = 0; r < 8; ++r)
                c[r] = bf2f(E2[e2base + (size_t)(q0 + r + 8 * hi) * SEQ + key]);
            BF16x16 bk0, bk1;                    // K tile as B-matrix (d-contig)
            const unsigned short* Kr = Kbf + ((size_t)bh * SEQ + key) * DK;
            bk0.q[0] = *(const uint4*)(Kr + hi * 16);
            bk0.q[1] = *(const uint4*)(Kr + hi * 16 + 8);
            bk1.q[0] = *(const uint4*)(Kr + 32 + hi * 16);
            bk1.q[1] = *(const uint4*)(Kr + 32 + hi * 16 + 8);
            c = wmma_bf16(a0.v, bk0.v, c);
            c = wmma_bf16(a1.v, bk1.v, c);
            #pragma unroll
            for (int r = 0; r < 8; ++r) {
                if (mask[mbase + (size_t)(q0 + r + 8 * hi) * SEQ + key])
                    c[r] = -1.0e8f;
                s[sub][r] = c[r];
            }
        }
        // online softmax: reductions across the 16 lanes of each half (wave32)
        #pragma unroll
        for (int r = 0; r < 8; ++r) {
            float mx = fmaxf(s[0][r], s[1][r]);
            #pragma unroll
            for (int d = 1; d < 16; d <<= 1) mx = fmaxf(mx, __shfl_xor(mx, d, 32));
            float newm = fmaxf(mrow[r], mx);
            float sc = __expf(mrow[r] - newm);
            float p0 = __expf(s[0][r] - newm);
            float p1 = __expf(s[1][r] - newm);
            float sum = p0 + p1;
            #pragma unroll
            for (int d = 1; d < 16; d <<= 1) sum += __shfl_xor(sum, d, 32);
            lrow[r] = lrow[r] * sc + sum;
            mrow[r] = newm;
            #pragma unroll
            for (int t = 0; t < 4; ++t) o[t][r] *= sc;
            Plds[w][r + 8 * hi][lo]      = f2bf(p0);   // C-layout -> LDS
            Plds[w][r + 8 * hi][16 + lo] = f2bf(p1);
        }
        asm volatile("s_wait_dscnt 0x0" ::: "memory"); // LDS stores visible

        BF16x16 ap;                               // P in A-matrix layout
        ap.q[0] = *(const uint4*)&Plds[w][lo][hi * 16];
        ap.q[1] = *(const uint4*)&Plds[w][lo][hi * 16 + 8];
        #pragma unroll
        for (int t = 0; t < 4; ++t) {             // V^T: contraction contiguous
            BF16x16 bv;
            const unsigned short* Vr =
                Vt + ((size_t)bh * DK + t * 16 + lo) * SEQ + kbase + hi * 16;
            bv.q[0] = *(const uint4*)(Vr);
            bv.q[1] = *(const uint4*)(Vr + 8);
            o[t] = wmma_bf16(ap.v, bv.v, o[t]);
        }
    }

    // normalize and store out[b, s, h*64+d] (fp32)
    #pragma unroll
    for (int r = 0; r < 8; ++r) {
        float inv = 1.0f / lrow[r];
        int row = q0 + r + 8 * hi;
        size_t obase = (size_t)b * SEQ * DMODEL + (size_t)row * DMODEL + h * DK;
        #pragma unroll
        for (int t = 0; t < 4; ++t)
            out[obase + t * 16 + lo] = o[t][r] * inv;
    }
}

// ---------------------------------------------------------------------------
extern "C" void kernel_launch(void* const* d_in, const int* in_sizes, int n_in,
                              void* d_out, int out_size, void* d_ws, size_t ws_size,
                              hipStream_t stream) {
    const float* query = (const float*)d_in[0];
    const float* key   = (const float*)d_in[1];
    const float* value = (const float*)d_in[2];
    const float* rel   = (const float*)d_in[3];
    const float* Wq    = (const float*)d_in[4];
    const float* bq    = (const float*)d_in[5];
    const float* Wk    = (const float*)d_in[6];
    const float* bk    = (const float*)d_in[7];
    const float* Wv    = (const float*)d_in[8];
    const float* bv    = (const float*)d_in[9];
    const unsigned char* mask = (const unsigned char*)d_in[10];
    float* out = (float*)d_out;

    // workspace layout (raw bf16 bits): Q 16MB | K 16MB | V^T 16MB | E2 256MB
    unsigned short* Qbf = (unsigned short*)d_ws;
    unsigned short* Kbf = Qbf + (size_t)BH * SEQ * DK;
    unsigned short* Vtw = Kbf + (size_t)BH * SEQ * DK;
    unsigned short* E2  = Vtw + (size_t)BH * SEQ * DK;

    dim3 pgrid(64, 16);   // 8192/128 m-blocks, 1024/64 n-blocks
    proj_gemm_kernel<<<pgrid, 256, 0, stream>>>(query, Wq, bq, Qbf, 0, 1.0f);
    proj_gemm_kernel<<<pgrid, 256, 0, stream>>>(key,   Wk, bk, Kbf, 0, 0.125f);
    proj_gemm_kernel<<<pgrid, 256, 0, stream>>>(value, Wv, bv, Vtw, 1, 1.0f);
    e2_kernel<<<dim3(16, 1024), 256, 0, stream>>>(Qbf, rel, E2);
    attn_kernel<<<1024, 256, 0, stream>>>(Qbf, Kbf, Vtw, E2, mask, out);

    (void)in_sizes; (void)n_in; (void)out_size; (void)ws_size;
}
